// TEPRNN2_35261681500186
// MI455X (gfx1250) — matrix operand order repaired and checked by reading, not compiled, not run on hardware
//
#include <hip/hip_runtime.h>
#include <hip/hip_bf16.h>

typedef __attribute__((ext_vector_type(16))) __bf16 v16bf;
typedef __attribute__((ext_vector_type(8)))  __bf16 v8bf;
typedef __attribute__((ext_vector_type(8)))  float  v8f;

#define SEQ  256
#define HDIM 64
#define ROWS_PER_BLOCK 32
#define M_TILES 2                 // 32 rows / 16
#define HSTRIDE 72                // padded bf16 elems per row (144B, 16B aligned, bank-spread)
#define NTHREADS 128

// gfx1250 hardware tanh. i/f/o gate weights are pre-scaled by 0.5 so
// sigmoid(x) = 0.5*tanh(acc) + 0.5 where acc = 0.5*x comes out of the WMMA.
#if __has_builtin(__builtin_amdgcn_tanhf)
__device__ __forceinline__ float fast_tanh(float x) {
    return __builtin_amdgcn_tanhf(x);
}
__device__ __forceinline__ float sigmoid_pre(float halfx) {   // input = 0.5*x
    return __builtin_fmaf(0.5f, __builtin_amdgcn_tanhf(halfx), 0.5f);
}
#else
__device__ __forceinline__ float fast_tanh(float x) {
    float e = __builtin_amdgcn_exp2f(2.8853900817779268f * x);
    return 1.0f - 2.0f * __builtin_amdgcn_rcpf(e + 1.0f);
}
__device__ __forceinline__ float sigmoid_pre(float halfx) {   // 1/(1+2^(-2*log2e*halfx))
    float e = __builtin_amdgcn_exp2f(-2.8853900817779268f * halfx);
    return __builtin_amdgcn_rcpf(1.0f + e);
}
#endif

__global__ __launch_bounds__(NTHREADS) void lstm_wmma_kernel(
    const float* __restrict__ x,     // (B, S)
    const float* __restrict__ W1,    // (4, 1)
    const float* __restrict__ b1,    // (4,)
    const float* __restrict__ W_ih,  // (256, 4)
    const float* __restrict__ W_hh,  // (256, 64)
    const float* __restrict__ b_ih,  // (256,)
    const float* __restrict__ b_hh,  // (256,)
    const float* __restrict__ W2,    // (1, 64)
    const float* __restrict__ b2,    // (1,)
    float* __restrict__ out)         // (B, 1)
{
    __shared__ __align__(16) __bf16 hbuf[2][ROWS_PER_BLOCK][HSTRIDE]; // h(t) exchange, double buffered

    const int tid  = threadIdx.x;
    const int w    = tid >> 5;      // wave id 0..3 -> hidden cols [w*16, w*16+16)
    const int lane = tid & 31;
    const int lhi  = lane >> 4;     // lane half (0/1)
    const int llo  = lane & 15;
    const bool loLane = (lane < 16);
    const float loMask = loLane ? 1.0f : 0.0f;
    const int rowBase = blockIdx.x * ROWS_PER_BLOCK;

    // ---- zero h(0) ----
    {
        __bf16* hz = &hbuf[0][0][0];
        for (int i = tid; i < ROWS_PER_BLOCK * HSTRIDE; i += NTHREADS) hz[i] = (__bf16)0.0f;
    }

    // ---- preload B fragments into registers ----
    // gates = [h | fc1 | 1] (16x96) x B (96x256).  B chunk0/1 = W_hh^T, chunk2 = W_ih^T + bias row.
    // 32x16 bf16 B layout: lane holds column N = llo; elements e -> K = lhi*16 + e (within chunk).
    // Gates i(0), f(1), o(3) pre-scaled by 0.5 (exact in bf16) to fold the sigmoid half-scale.
    v16bf Bf[4][2];   // [gate][kchunk 0/1]
    v16bf B2f[4];     // [gate] chunk2
    #pragma unroll
    for (int g = 0; g < 4; ++g) {
        const float gs = (g == 2) ? 1.0f : 0.5f;
        const int gcol = (g * 4 + w) * 16 + llo;   // global gate column (N)
        #pragma unroll
        for (int kk = 0; kk < 2; ++kk) {
            const int kb = kk * 32 + lhi * 16;
            v16bf t;
            #pragma unroll
            for (int e = 0; e < 16; ++e) t[e] = (__bf16)(gs * W_hh[gcol * HDIM + kb + e]);
            Bf[g][kk] = t;
        }
        v16bf t2;
        #pragma unroll
        for (int e = 0; e < 16; ++e) t2[e] = (__bf16)0.0f;
        if (loLane) {               // K_local 0..7 live only in lane half 0
            #pragma unroll
            for (int j = 0; j < 4; ++j) t2[j] = (__bf16)(gs * W_ih[gcol * 4 + j]);
            t2[4] = (__bf16)(gs * (b_ih[gcol] + b_hh[gcol]));  // "1"-row -> bias folded into matmul
        }
        B2f[g] = t2;
    }

    // per-lane scalar params
    float w1r[4], b1r[4];
    #pragma unroll
    for (int j = 0; j < 4; ++j) { w1r[j] = W1[j]; b1r[j] = b1[j]; }

    // cell state: D-layout (lane = hidden col llo of slice w, elem e -> row lhi*8+e)
    v8f cfrag[M_TILES];
    #pragma unroll
    for (int mt = 0; mt < M_TILES; ++mt) {
        #pragma unroll
        for (int e = 0; e < 8; ++e) cfrag[mt][e] = 0.0f;
    }

    // A chunk2 persistent fragments: elements 4..15 are loop-invariant
    // ([4] = 1.0 indicator on lane half 0, rest zero); only [0..3] (fc1) update per step.
    v16bf a2v[M_TILES];
    #pragma unroll
    for (int mt = 0; mt < M_TILES; ++mt) {
        #pragma unroll
        for (int e = 0; e < 16; ++e) a2v[mt][e] = (__bf16)0.0f;
        a2v[mt][4] = loLane ? (__bf16)1.0f : (__bf16)0.0f;
    }

    // x prefetch (one step ahead), per-mt base pointers; last-step prefetch wraps
    // to offset 0 (in bounds, value unused).
    const float* xp[M_TILES];
    float xc[M_TILES], xn[M_TILES];
    #pragma unroll
    for (int mt = 0; mt < M_TILES; ++mt) {
        xp[mt] = x + (size_t)(rowBase + mt * 16 + llo) * SEQ;
        xc[mt] = xp[mt][0];
    }

    __syncthreads();

    int cur = 0;
    #pragma unroll 1
    for (int t = 0; t < SEQ; ++t) {
        const int tn = (t + 1) & (SEQ - 1);     // wrap instead of clamp: 1 s_and, no select
        #pragma unroll
        for (int mt = 0; mt < M_TILES; ++mt)
            xn[mt] = xp[mt][tn];

        const int nxt = cur ^ 1;

        #pragma unroll
        for (int mt = 0; mt < M_TILES; ++mt) {
            // ---- A fragments (16x32 bf16 layout: lane = row llo; elems 0..15 -> K=lhi*16+0..15)
            //      from h(t-1) in LDS ----
            const __bf16* hrow = &hbuf[cur][mt * 16 + llo][0];
            const int kb = lhi * 16;
            v8bf a0lo = *(const v8bf*)(hrow + kb);
            v8bf a0hi = *(const v8bf*)(hrow + kb + 8);
            v8bf a1lo = *(const v8bf*)(hrow + kb + 32);
            v8bf a1hi = *(const v8bf*)(hrow + kb + 40);
            v16bf a0 = __builtin_shufflevector(a0lo, a0hi, 0,1,2,3,4,5,6,7,8,9,10,11,12,13,14,15);
            v16bf a1 = __builtin_shufflevector(a1lo, a1hi, 0,1,2,3,4,5,6,7,8,9,10,11,12,13,14,15);

            // fc1(x_t) -> a2 elements 0..3; mask lanes>=16 by 0/1 float mul (dual-issuable),
            // then pair conversions lower to v_cvt_pk_bf16_f32 into a2v's first two VGPRs.
            {
                const float xv = xc[mt];
                const float m0 = loMask * fast_tanh(__builtin_fmaf(xv, w1r[0], b1r[0]));
                const float m1 = loMask * fast_tanh(__builtin_fmaf(xv, w1r[1], b1r[1]));
                const float m2 = loMask * fast_tanh(__builtin_fmaf(xv, w1r[2], b1r[2]));
                const float m3 = loMask * fast_tanh(__builtin_fmaf(xv, w1r[3], b1r[3]));
                a2v[mt][0] = (__bf16)m0;
                a2v[mt][1] = (__bf16)m1;
                a2v[mt][2] = (__bf16)m2;
                a2v[mt][3] = (__bf16)m3;
            }

            // ---- gates: 4 independent chains of 3 WMMAs ----
            v8f acc[4];
            #pragma unroll
            for (int g = 0; g < 4; ++g) {
                v8f z = {};
                z = __builtin_amdgcn_wmma_f32_16x16x32_bf16(false, a0, false, Bf[g][0], (short)0, z, false, false);
                z = __builtin_amdgcn_wmma_f32_16x16x32_bf16(false, a1, false, Bf[g][1], (short)0, z, false, false);
                z = __builtin_amdgcn_wmma_f32_16x16x32_bf16(false, a2v[mt], false, B2f[g], (short)0, z, false, false);
                acc[g] = z;
            }

            // ---- lane-local LSTM cell update (i/f/o accs are pre-halved) ----
            #pragma unroll
            for (int e = 0; e < 8; ++e) {
                const float ig = sigmoid_pre(acc[0][e]);
                const float fg = sigmoid_pre(acc[1][e]);
                const float gg = fast_tanh(acc[2][e]);
                const float og = sigmoid_pre(acc[3][e]);
                const float cc = __builtin_fmaf(fg, cfrag[mt][e], ig * gg);
                cfrag[mt][e] = cc;
                const float hh = og * fast_tanh(cc);
                const int row = mt * 16 + lhi * 8 + e;
                hbuf[nxt][row][w * 16 + llo] = (__bf16)hh;      // D-layout -> row-major for next A load
            }
        }

        #pragma unroll
        for (int mt = 0; mt < M_TILES; ++mt) xc[mt] = xn[mt];
        __syncthreads();
        cur = nxt;
    }

    // ---- output projection: out[b] = h_last . W2 + b2 (h_last bf16 in hbuf[cur]) ----
    if (tid < ROWS_PER_BLOCK) {
        const __bf16* hr = &hbuf[cur][tid][0];
        float s = 0.0f;
        #pragma unroll
        for (int k = 0; k < HDIM; ++k) s = __builtin_fmaf((float)hr[k], W2[k], s);
        out[rowBase + tid] = s + b2[0];
    }
}

extern "C" void kernel_launch(void* const* d_in, const int* in_sizes, int n_in,
                              void* d_out, int out_size, void* d_ws, size_t ws_size,
                              hipStream_t stream) {
    const float* x    = (const float*)d_in[0];
    const float* W1   = (const float*)d_in[1];
    const float* b1   = (const float*)d_in[2];
    const float* W_ih = (const float*)d_in[3];
    const float* W_hh = (const float*)d_in[4];
    const float* b_ih = (const float*)d_in[5];
    const float* b_hh = (const float*)d_in[6];
    const float* W2   = (const float*)d_in[7];
    const float* b2   = (const float*)d_in[8];
    float* out = (float*)d_out;

    const int B = 4096;
    dim3 grid(B / ROWS_PER_BLOCK);   // 128 blocks
    dim3 block(NTHREADS);            // 4 waves: one 16-hidden-col slice each
    lstm_wmma_kernel<<<grid, block, 0, stream>>>(x, W1, b1, W_ih, W_hh, b_ih, b_hh, W2, b2, out);
}
